// AttentionHead_28406913696251
// MI455X (gfx1250) — compile-verified, matrix operand order checked by reading, not run on hardware
//
#include <hip/hip_runtime.h>

// ---------------------------------------------------------------------------
// AttentionHead for MI455X (gfx1250, wave32, WMMA bf16, async-to-LDS DMA)
//   pass 0: wt_prep — convert/transpose weights once: f32 [F][DK] -> bf16 [DK][F]
//   pass 1: qp/kp = X @ W, vpT = (V @ Wv)^T.  X: register-pipelined f32->bf16
//           staging; W: double-buffered global_load_async_to_lds_b128 DMA.
//   pass 2: flash-attention over 32-key blocks; K/V^T staged with async DMA
//           double-buffering (ASYNCcnt), WMMA bf16 for QK^T and PV.
// ---------------------------------------------------------------------------

typedef __bf16 v16bf __attribute__((ext_vector_type(16)));
typedef float  v8f   __attribute__((ext_vector_type(8)));
typedef unsigned int u32x4 __attribute__((ext_vector_type(4)));

union BF16Frag { v16bf v; u32x4 u[2]; };
union BF16Pack4 { __bf16 h[4]; unsigned long long q; };

#define B_DIM  8
#define S_DIM  2048
#define F_DIM  1024
#define DK_DIM 128
#define M_TOT  (B_DIM * S_DIM)          // 16384 rows in projection GEMMs
#define DKS    (DK_DIM * S_DIM)         // 262144, one batch of vpT

// Async DMA: global -> LDS, 16B per lane, tracked by ASYNCcnt (CDNA5).
// GVS addressing: mem = SGPR64 + VGPR32 + ioffset. LDS dest = low 32 bits of
// the generic pointer (ISA 10.2).
__device__ __forceinline__ void async_copy_b128(unsigned lds_addr,
                                                unsigned voff,
                                                const void* sbase) {
  asm volatile("global_load_async_to_lds_b128 %0, %1, %2"
               :: "v"(lds_addr), "v"(voff), "s"(sbase) : "memory");
}

// ---------------------------------------------------------------------------
// Weight prep: wt[d][f] = (bf16) w[f][d].  One-time 0.5MB transpose/convert.
// ---------------------------------------------------------------------------
__global__ __launch_bounds__(256) void wt_prep(const float* __restrict__ w,
                                               __bf16* __restrict__ wt)
{
  const int tid = threadIdx.x;
  #pragma unroll
  for (int p = 0; p < 8; ++p) {
    const int e = blockIdx.x * 2048 + p * 256 + tid;   // 0..131071, coalesced
    const int f = e >> 7;
    const int d = e & 127;
    wt[(size_t)d * F_DIM + f] = (__bf16)w[e];
  }
}

// ---------------------------------------------------------------------------
// Projection: out = X @ W (16384x1024 @ 1024x128). 64 m-rows per block,
// 8 waves = 4 m-strips x 2 d-halves, each wave 16m x 64d (4 accumulators).
// X loads register-pipelined one k-step ahead; W DMA'd one k-step ahead.
// tmode=1 stores transposed [B][DK][S].
// ---------------------------------------------------------------------------
__global__ __launch_bounds__(256) void proj_bf16_wmma(
    const float* __restrict__ x, const __bf16* __restrict__ wt,  // wt: [DK][F]
    __bf16* __restrict__ out, const int tmode)
{
  __shared__ __bf16 sX [64 * 32];        // [m_local][f_local]   4KB
  __shared__ __bf16 sWT[2][128 * 32];    // [d][f_local] x2     16KB

  const int tid    = threadIdx.x;
  const int wave   = tid >> 5;
  const int lane   = tid & 31;
  const int col16  = lane & 15;
  const int fhalf  = (lane >> 4) << 3;
  const int m0     = blockIdx.x * 64;
  const int mstrip = wave >> 1;          // 0..3
  const int dbase  = (wave & 1) * 4;     // d-tile offset: 0 or 4

  const unsigned sWT_lds[2] = { (unsigned)(uintptr_t)&sWT[0][0],
                                (unsigned)(uintptr_t)&sWT[1][0] };

  float4 xreg[2];                        // X staging regs (64x32 f32, 2/thread)
  auto load_x = [&](int kb) {
    #pragma unroll
    for (int p = 0; p < 2; ++p) {
      const int idx = tid + 256 * p;
      xreg[p] = *(const float4*)&x[(size_t)(m0 + (idx >> 3)) * F_DIM + kb +
                                   ((idx & 7) << 2)];
    }
  };
  // W block [128 d][32 f] = 512 x 16B chunks, 2 per thread, straight DMA
  auto stage_w = [&](int kb, int buf) {
    #pragma unroll
    for (int i = 0; i < 2; ++i) {
      const int c = tid + 256 * i;       // chunk 0..511
      const int d = c >> 2;
      async_copy_b128(sWT_lds[buf] + c * 16,
                      (unsigned)(d * 2048 + kb * 2 + (c & 3) * 16), wt);
    }
  };

  v8f acc[4] = {};
  load_x(0);
  stage_w(0, 0);

  int wbuf = 0;
  for (int kb = 0; kb < F_DIM; kb += 32) {
    // ---- commit staged X registers to LDS (f32 -> bf16, b64 stores) ------
    #pragma unroll
    for (int p = 0; p < 2; ++p) {
      const int idx = tid + 256 * p;
      BF16Pack4 pk;
      pk.h[0] = (__bf16)xreg[p].x; pk.h[1] = (__bf16)xreg[p].y;
      pk.h[2] = (__bf16)xreg[p].z; pk.h[3] = (__bf16)xreg[p].w;
      *(unsigned long long*)&sX[(idx >> 3) * 32 + ((idx & 7) << 2)] = pk.q;
    }
    // ---- DMA next W block, wait for current one ---------------------------
    if (kb + 32 < F_DIM) {
      stage_w(kb + 32, wbuf ^ 1);
      asm volatile("s_wait_asynccnt 2" ::: "memory");
    } else {
      asm volatile("s_wait_asynccnt 0" ::: "memory");
    }
    __syncthreads();

    // ---- issue next k-step's X loads (in flight during WMMAs) ------------
    if (kb + 32 < F_DIM) load_x(kb + 32);

    // ---- compute: A frag shared per m-strip, 4 d-tile WMMAs --------------
    BF16Frag a;
    const __bf16* arow = &sX[(mstrip * 16 + col16) * 32];
    a.u[0] = *(const u32x4*)&arow[fhalf];
    a.u[1] = *(const u32x4*)&arow[fhalf + 16];

    const __bf16* sW = sWT[wbuf];
    #pragma unroll
    for (int dt = 0; dt < 4; ++dt) {
      BF16Frag bfr;
      const __bf16* brow = &sW[((dbase + dt) * 16 + col16) * 32];
      bfr.u[0] = *(const u32x4*)&brow[fhalf];
      bfr.u[1] = *(const u32x4*)&brow[fhalf + 16];
      acc[dt] = __builtin_amdgcn_wmma_f32_16x16x32_bf16(
          false, a.v, false, bfr.v, (short)0, acc[dt], false, false);
    }
    __syncthreads();   // LDS consumed before next iteration overwrites it
    wbuf ^= 1;
  }

  const int hi = lane >> 4;
  if (!tmode) {
    #pragma unroll
    for (int dt = 0; dt < 4; ++dt)
      #pragma unroll
      for (int r = 0; r < 8; ++r) {
        const int m = m0 + mstrip * 16 + r + 8 * hi;
        out[(size_t)m * DK_DIM + (dbase + dt) * 16 + col16] = (__bf16)acc[dt][r];
      }
  } else {
    #pragma unroll
    for (int dt = 0; dt < 4; ++dt)
      #pragma unroll
      for (int r = 0; r < 8; ++r) {
        const int m = m0 + mstrip * 16 + r + 8 * hi;
        const int d = (dbase + dt) * 16 + col16;
        out[(size_t)(m >> 11) * DKS + (size_t)d * S_DIM + (m & 2047)] =
            (__bf16)acc[dt][r];
      }
  }
}

// ---------------------------------------------------------------------------
// Flash attention: 128 threads (4 wave32), 64 queries per block, grid 256.
// K block [32][128] and V^T block [128][32] double-buffered via async DMA.
// ---------------------------------------------------------------------------
__global__ __launch_bounds__(128) void flash_attn_wmma(
    const __bf16* __restrict__ qp, const __bf16* __restrict__ kp,
    const __bf16* __restrict__ vpT, float* __restrict__ out)
{
  __shared__ __bf16 sK [2][32 * 128];   // [key][d]
  __shared__ __bf16 sVT[2][128 * 32];   // [d][key]
  __shared__ __bf16 sP [4][16 * 32];    // per-wave P tile [q][key]

  const int tid   = threadIdx.x;
  const int wave  = tid >> 5;
  const int lane  = tid & 31;
  const int col16 = lane & 15;
  const int fhalf = (lane >> 4) << 3;
  const int hi    = lane >> 4;

  const int b  = blockIdx.x >> 5;               // 32 q-blocks per batch
  const int q0 = (blockIdx.x & 31) * 64;
  const size_t base = (size_t)b * S_DIM * DK_DIM;
  const __bf16* kpB = kp + base;                // [s][d] for this batch
  const __bf16* vtB = vpT + (size_t)b * DKS;    // [d][s] for this batch

  const unsigned sK_lds[2]  = { (unsigned)(uintptr_t)&sK[0][0],
                                (unsigned)(uintptr_t)&sK[1][0] };
  const unsigned sVT_lds[2] = { (unsigned)(uintptr_t)&sVT[0][0],
                                (unsigned)(uintptr_t)&sVT[1][0] };

  // loop-invariant Q fragments: 4 K-steps covering d=0..127
  const int qrow = q0 + wave * 16 + col16;
  BF16Frag aq[4];
  #pragma unroll
  for (int kk = 0; kk < 4; ++kk) {
    const __bf16* src = &qp[base + (size_t)qrow * DK_DIM + kk * 32 + fhalf];
    aq[kk].u[0] = *(const u32x4*)src;
    aq[kk].u[1] = *(const u32x4*)(src + 16);
  }

  v8f oacc[8] = {};
  float m_run[8], l_run[8];
  #pragma unroll
  for (int r = 0; r < 8; ++r) { m_run[r] = -3.0e30f; l_run[r] = 0.0f; }

  const float scale = 0.08838834764831845f;     // 1/sqrt(128)

  // ---- async stage of one 32-key block: 512 + 512 b128 chunks, 8/thread ---
  auto stage_kv = [&](int k0, int buf) {
    #pragma unroll
    for (int i = 0; i < 4; ++i) {               // K block: linear copy
      const int c = tid + 128 * i;              // chunk 0..511 (16B each)
      async_copy_b128(sK_lds[buf] + c * 16, (unsigned)(k0 * 256 + c * 16), kpB);
    }
    #pragma unroll
    for (int i = 0; i < 4; ++i) {               // V^T block: rows of 64B
      const int c = tid + 128 * i;
      const int d = c >> 2;
      async_copy_b128(sVT_lds[buf] + c * 16,
                      (unsigned)(d * 4096 + k0 * 2 + (c & 3) * 16), vtB);
    }
  };

  stage_kv(0, 0);                               // prologue: block 0

  for (int j = 0; j < S_DIM / 32; ++j) {
    const int cur = j & 1;
    if (j + 1 < S_DIM / 32) {
      stage_kv((j + 1) * 32, cur ^ 1);          // DMA next block
      asm volatile("s_wait_asynccnt 8" ::: "memory");  // cur's 8 ops done
    } else {
      asm volatile("s_wait_asynccnt 0" ::: "memory");
    }
    __syncthreads();

    const __bf16* sKc  = sK[cur];
    const __bf16* sVTc = sVT[cur];

    // ---- scores: S[16q x 32k], K=DK via 4 WMMA steps per key sub-tile ----
    v8f s0 = {}, s1 = {};
    #pragma unroll
    for (int kk = 0; kk < 4; ++kk) {
      BF16Frag bk;
      const __bf16* krow = &sKc[col16 * 128 + kk * 32 + fhalf];
      bk.u[0] = *(const u32x4*)krow;
      bk.u[1] = *(const u32x4*)(krow + 16);
      s0 = __builtin_amdgcn_wmma_f32_16x16x32_bf16(
          false, aq[kk].v, false, bk.v, (short)0, s0, false, false);
    }
    #pragma unroll
    for (int kk = 0; kk < 4; ++kk) {
      BF16Frag bk;
      const __bf16* krow = &sKc[(16 + col16) * 128 + kk * 32 + fhalf];
      bk.u[0] = *(const u32x4*)krow;
      bk.u[1] = *(const u32x4*)(krow + 16);
      s1 = __builtin_amdgcn_wmma_f32_16x16x32_bf16(
          false, aq[kk].v, false, bk.v, (short)0, s1, false, false);
    }

    // ---- online softmax (per-lane stats match C/D row layout) ------------
    float p0[8], p1[8], alpha[8];
    #pragma unroll
    for (int r = 0; r < 8; ++r) {
      const float a0 = s0[r] * scale;
      const float a1 = s1[r] * scale;
      float rmax = fmaxf(a0, a1);
      #pragma unroll
      for (int off = 8; off >= 1; off >>= 1)    // stays within each 16-half
        rmax = fmaxf(rmax, __shfl_xor(rmax, off, 32));
      const float m_new = fmaxf(m_run[r], rmax);
      alpha[r] = __expf(m_run[r] - m_new);
      m_run[r] = m_new;
      p0[r] = __expf(a0 - m_new);
      p1[r] = __expf(a1 - m_new);
      float psum = p0[r] + p1[r];
      #pragma unroll
      for (int off = 8; off >= 1; off >>= 1)
        psum += __shfl_xor(psum, off, 32);
      l_run[r] = l_run[r] * alpha[r] + psum;
    }
    #pragma unroll
    for (int dt = 0; dt < 8; ++dt)
      #pragma unroll
      for (int r = 0; r < 8; ++r)
        oacc[dt][r] *= alpha[r];

    // ---- P (C-layout) -> LDS -> re-read in A layout ----------------------
    __bf16* pw = sP[wave];
    #pragma unroll
    for (int r = 0; r < 8; ++r) {
      const int prow = r + 8 * hi;
      pw[prow * 32 + col16]      = (__bf16)p0[r];
      pw[prow * 32 + 16 + col16] = (__bf16)p1[r];
    }
    asm volatile("s_wait_dscnt 0" ::: "memory");  // same-wave LDS RAW fence

    BF16Frag pa;
    pa.u[0] = *(const u32x4*)&pw[col16 * 32 + fhalf];
    pa.u[1] = *(const u32x4*)&pw[col16 * 32 + fhalf + 16];

    // ---- O += P @ V : 8 d-tiles, K=32 keys -------------------------------
    #pragma unroll
    for (int dt = 0; dt < 8; ++dt) {
      BF16Frag bv;
      const __bf16* vrow = &sVTc[(dt * 16 + col16) * 32];
      bv.u[0] = *(const u32x4*)&vrow[fhalf];
      bv.u[1] = *(const u32x4*)&vrow[fhalf + 16];
      oacc[dt] = __builtin_amdgcn_wmma_f32_16x16x32_bf16(
          false, pa.v, false, bv.v, (short)0, oacc[dt], false, false);
    }
    __syncthreads();   // all waves done with 'cur' before it is re-DMA'd
  }

  // ---- normalize and store f32 output ------------------------------------
  #pragma unroll
  for (int r = 0; r < 8; ++r) {
    const float inv = 1.0f / l_run[r];
    const int qg = q0 + wave * 16 + r + 8 * hi;
    #pragma unroll
    for (int dt = 0; dt < 8; ++dt)
      out[base + (size_t)qg * DK_DIM + dt * 16 + col16] = oacc[dt][r] * inv;
  }
}

// ---------------------------------------------------------------------------
extern "C" void kernel_launch(void* const* d_in, const int* in_sizes, int n_in,
                              void* d_out, int out_size, void* d_ws, size_t ws_size,
                              hipStream_t stream) {
  const float* q  = (const float*)d_in[0];
  const float* k  = (const float*)d_in[1];
  const float* v  = (const float*)d_in[2];
  const float* wq = (const float*)d_in[3];
  const float* wk = (const float*)d_in[4];
  const float* wv = (const float*)d_in[5];
  float* out = (float*)d_out;

  const size_t projElems = (size_t)M_TOT * DK_DIM;    // 2M bf16 each
  const size_t wtElems   = (size_t)DK_DIM * F_DIM;    // 128K bf16 each
  __bf16* qp  = (__bf16*)d_ws;
  __bf16* kp  = qp  + projElems;
  __bf16* vpT = kp  + projElems;
  __bf16* wqT = vpT + projElems;
  __bf16* wkT = wqT + wtElems;
  __bf16* wvT = wkT + wtElems;

  dim3 bl(256);
  wt_prep<<<dim3(64), bl, 0, stream>>>(wq, wqT);
  wt_prep<<<dim3(64), bl, 0, stream>>>(wk, wkT);
  wt_prep<<<dim3(64), bl, 0, stream>>>(wv, wvT);

  dim3 gproj(M_TOT / 64);
  proj_bf16_wmma<<<gproj, bl, 0, stream>>>(q, wqT, qp,  0);
  proj_bf16_wmma<<<gproj, bl, 0, stream>>>(k, wkT, kp,  0);
  proj_bf16_wmma<<<gproj, bl, 0, stream>>>(v, wvT, vpT, 1);   // transposed

  dim3 gattn(B_DIM * (S_DIM / 64)), battn(128);
  flash_attn_wmma<<<gattn, battn, 0, stream>>>(qp, kp, vpT, out);
}
